// SkipGRU_22204980920798
// MI455X (gfx1250) — matrix operand order repaired and checked by reading, not compiled, not run on hardware
//
#include <hip/hip_runtime.h>
#include <cstddef>

// SkipGRU for MI455X (gfx1250).
// Matrix math via V_WMMA_F32_16X16X32_F16 with two-term f16 splitting
// (hi/lo), giving ~fp32 accuracy at f16-WMMA throughput:
//   A*B ~= hiA*hiB + loA*hiB + hiA*loB   (3 WMMAs per 16x16x32 tile)
// B=128, T=512, C=H=512, PT=16, SKIP=32 -> recurrent batch N=4096, 16 steps.

typedef _Float16 v4h __attribute__((ext_vector_type(4)));
typedef _Float16 v8h __attribute__((ext_vector_type(8)));
typedef _Float16 v16h __attribute__((ext_vector_type(16)));
typedef float v4f __attribute__((ext_vector_type(4)));
typedef float v8f __attribute__((ext_vector_type(8)));

#define GRU_H      512
#define GRU_3H     1536
#define GRU_ROWS   4096   // B*SKIP
#define GRU_T      16     // PT
#define TILE_M     64
#define TILE_H     32
#define TILE_K     32
// LDS strides in halves. 40 halves = 20 dwords; m*20 mod 64 distinct
// multiples of 4 for m=0..15 -> each half-wave of a b128 load covers all 16
// bank-quads exactly once (optimal two-pass for 32 lanes).
#define SAK 40
#define SBK 40

#define WMMA_F16(ACC, A, B) \
  ACC = __builtin_amdgcn_wmma_f32_16x16x32_f16(false, (A), false, (B), \
                                               (short)0, (ACC), false, false)

__device__ __forceinline__ float sigmoid_f(float x) {
  return 1.0f / (1.0f + __expf(-x));
}

__device__ __forceinline__ v16h cat8(v8h a, v8h b) {
  return __builtin_shufflevector(a, b, 0, 1, 2, 3, 4, 5, 6, 7,
                                 8, 9, 10, 11, 12, 13, 14, 15);
}

__global__ void zero_f32(float* __restrict__ p, int n) {
  int i = (blockIdx.x * blockDim.x + threadIdx.x) * 4;
  if (i < n) {
    v4f z = {0.f, 0.f, 0.f, 0.f};
    *(v4f*)(p + i) = z;
  }
}

// One-time weight split: W/U (f32) -> hi/lo f16 pair.
__global__ void split_f16(const float* __restrict__ src,
                          _Float16* __restrict__ hi,
                          _Float16* __restrict__ lo, int n) {
  int i = blockIdx.x * blockDim.x + threadIdx.x;
  if (i < n) {
    float v = src[i];
    _Float16 h = (_Float16)v;
    hi[i] = h;
    lo[i] = (_Float16)(v - (float)h);
  }
}

__global__ __launch_bounds__(256) void skipgru_step(
    const float* __restrict__ x,        // (128, 512, 512)
    const _Float16* __restrict__ Whi,   // (512, 1536) [z|r|h]
    const _Float16* __restrict__ Wlo,
    const _Float16* __restrict__ Uhi,
    const _Float16* __restrict__ Ulo,
    const float* __restrict__ bias,     // (2, 1536)
    const float* __restrict__ hprev,    // (4096, 512)
    float* __restrict__ hout,           // (4096, 512)
    int t) {
  __shared__ _Float16 sAh[TILE_M * SAK];       // A hi, [m][K]
  __shared__ _Float16 sAl[TILE_M * SAK];       // A lo
  __shared__ _Float16 sBh[3 * TILE_H * SBK];   // B hi, transposed [gate][n][K]
  __shared__ _Float16 sBl[3 * TILE_H * SBK];   // B lo

  const int tid  = threadIdx.x;
  const int lane = tid & 31;
  const int wv   = tid >> 5;       // wave 0..7
  const int mi   = wv & 3;         // M subtile (16 rows)
  const int ni   = wv >> 2;        // N subtile (16 cols)
  const int g    = lane >> 4;      // half-wave
  const int lm   = lane & 15;

  const int rowBase = blockIdx.x * TILE_M;  // recurrent-row block
  const int hcol0   = blockIdx.y * TILE_H;  // hidden-column block

  // staging: 8 threads per 32-wide row, one 4-element chunk each
  const int rr = tid >> 3;         // 0..31
  const int c4 = (tid & 7) << 2;   // 0,4,...,28

  v8f accz  = {0.f,0.f,0.f,0.f,0.f,0.f,0.f,0.f};  // z: xW + hU (merged)
  v8f accr  = {0.f,0.f,0.f,0.f,0.f,0.f,0.f,0.f};  // r: xW + hU (merged)
  v8f accxh = {0.f,0.f,0.f,0.f,0.f,0.f,0.f,0.f};  // cand, input path
  v8f accrh = {0.f,0.f,0.f,0.f,0.f,0.f,0.f,0.f};  // cand, recurrent path

#pragma unroll
  for (int phase = 0; phase < 2; ++phase) {
    const _Float16* __restrict__ Bh = (phase == 0) ? Whi : Uhi;
    const _Float16* __restrict__ Bl = (phase == 0) ? Wlo : Ulo;
    for (int kt = 0; kt < GRU_H / TILE_K; ++kt) {
      const int kbase = kt * TILE_K;
      __syncthreads();
      // ---- stage A tile (64 x 32 f32 -> hi/lo f16) ----
      {
        const int n0 = rowBase + rr;
        const int n1 = n0 + 32;
        const float* p0;
        const float* p1;
        if (phase == 0) {
          // s_t row n lives at x[b = n>>5, time = t*32 + (n&31), :]
          p0 = x + ((size_t)(n0 >> 5) * 512 + (size_t)(t * 32 + (n0 & 31))) * 512
                 + kbase + c4;
          p1 = x + ((size_t)(n1 >> 5) * 512 + (size_t)(t * 32 + (n1 & 31))) * 512
                 + kbase + c4;
        } else {
          p0 = hprev + (size_t)n0 * GRU_H + kbase + c4;
          p1 = hprev + (size_t)n1 * GRU_H + kbase + c4;
        }
        v4f va0 = *(const v4f*)p0;
        v4f va1 = *(const v4f*)p1;
        v4h h0v, l0v, h1v, l1v;
#pragma unroll
        for (int i = 0; i < 4; ++i) {
          h0v[i] = (_Float16)va0[i]; l0v[i] = (_Float16)(va0[i] - (float)h0v[i]);
          h1v[i] = (_Float16)va1[i]; l1v[i] = (_Float16)(va1[i] - (float)h1v[i]);
        }
        *(v4h*)&sAh[rr * SAK + c4]        = h0v;
        *(v4h*)&sAl[rr * SAK + c4]        = l0v;
        *(v4h*)&sAh[(rr + 32) * SAK + c4] = h1v;
        *(v4h*)&sAl[(rr + 32) * SAK + c4] = l1v;
      }
      // ---- stage B tiles (pre-split f16), transposed to [n][K] ----
#pragma unroll
      for (int gate = 0; gate < 3; ++gate) {
        const size_t goff = (size_t)(kbase + rr) * GRU_3H + gate * GRU_H
                            + hcol0 + c4;
        v4h h4 = *(const v4h*)(Bh + goff);
        v4h l4 = *(const v4h*)(Bl + goff);
#pragma unroll
        for (int i = 0; i < 4; ++i) {
          sBh[(gate * TILE_H + c4 + i) * SBK + rr] = h4[i];
          sBl[(gate * TILE_H + c4 + i) * SBK + rr] = l4[i];
        }
      }
      __syncthreads();
      // ---- WMMA: 3 gates x (hi*hi + lo*hi + hi*lo) over K=32 ----
      // A layout (16-bit, 16x32): lane needs halves [8g, 8g+8) and
      // [16+8g, 16+8g+8) of its row -> two b128 LDS reads.
      const _Float16* arH = &sAh[(mi * 16 + lm) * SAK];
      const _Float16* arL = &sAl[(mi * 16 + lm) * SAK];
      v16h ah = cat8(*(const v8h*)&arH[8 * g], *(const v8h*)&arH[16 + 8 * g]);
      v16h al = cat8(*(const v8h*)&arL[8 * g], *(const v8h*)&arL[16 + 8 * g]);
      // B layout (32x16): lane needs its column's halves [8g, 8g+8) and
      // [16+8g, 16+8g+8) (same k order as A) -> two b128 LDS reads.
      const int bn = ni * 16 + lm;
#pragma unroll
      for (int gate = 0; gate < 3; ++gate) {
        const int base = (gate * TILE_H + bn) * SBK + 8 * g;
        v16h bh = cat8(*(const v8h*)&sBh[base], *(const v8h*)&sBh[base + 16]);
        v16h bl = cat8(*(const v8h*)&sBl[base], *(const v8h*)&sBl[base + 16]);
        if (gate == 0) {
          WMMA_F16(accz, ah, bh);
          WMMA_F16(accz, al, bh);
          WMMA_F16(accz, ah, bl);
        } else if (gate == 1) {
          WMMA_F16(accr, ah, bh);
          WMMA_F16(accr, al, bh);
          WMMA_F16(accr, ah, bl);
        } else if (phase == 0) {
          WMMA_F16(accxh, ah, bh);
          WMMA_F16(accxh, al, bh);
          WMMA_F16(accxh, ah, bl);
        } else {
          WMMA_F16(accrh, ah, bh);
          WMMA_F16(accrh, al, bh);
          WMMA_F16(accrh, ah, bl);
        }
      }
    }
  }

  // ---- gates + state update ----
  // f32 C/D layout: element v of each lane is (row = v + 8*(lane/16),
  // col = lane%16).
  const int j = hcol0 + ni * 16 + lm;                 // hidden column
  const float bz_c  = bias[j]              + bias[GRU_3H + j];
  const float br_c  = bias[GRU_H + j]      + bias[GRU_3H + GRU_H + j];
  const float bxh_c = bias[2 * GRU_H + j];
  const float brh_c = bias[GRU_3H + 2 * GRU_H + j];
#pragma unroll
  for (int v = 0; v < 8; ++v) {
    const int nrow = rowBase + mi * 16 + v + 8 * g;   // recurrent row
    const float hp = hprev[(size_t)nrow * GRU_H + j];
    const float zg = sigmoid_f(accz[v] + bz_c);
    const float rg = sigmoid_f(accr[v] + br_c);
    const float cand = accxh[v] + bxh_c + rg * (accrh[v] + brh_c);
    const float hh = cand > 0.0f ? cand : 0.0f;
    hout[(size_t)nrow * GRU_H + j] = zg * hp + (1.0f - zg) * hh;
  }
}

extern "C" void kernel_launch(void* const* d_in, const int* in_sizes, int n_in,
                              void* d_out, int out_size, void* d_ws,
                              size_t ws_size, hipStream_t stream) {
  const float* x = (const float*)d_in[0];
  const float* W = (const float*)d_in[1];
  const float* U = (const float*)d_in[2];
  const float* b = (const float*)d_in[3];
  float* out = (float*)d_out;   // 4096*512 f32 (== reference reshape)

  // Workspace layout: [h ping buffer 8 MB][Whi][Wlo][Uhi][Ulo] (f16, 1.5 MB ea)
  const int nstate = GRU_ROWS * GRU_H;        // 2M floats
  const int nw = GRU_H * GRU_3H;              // 786432 weights per matrix
  float* hws = (float*)d_ws;
  _Float16* Whi = (_Float16*)((char*)d_ws + (size_t)nstate * sizeof(float));
  _Float16* Wlo = Whi + nw;
  _Float16* Uhi = Wlo + nw;
  _Float16* Ulo = Uhi + nw;

  // h0 = 0: zero d_out, which is the t=0 read buffer in the ping-pong below.
  zero_f32<<<(nstate / 4 + 255) / 256, 256, 0, stream>>>(out, nstate);
  split_f16<<<(nw + 255) / 256, 256, 0, stream>>>(W, Whi, Wlo, nw);
  split_f16<<<(nw + 255) / 256, 256, 0, stream>>>(U, Uhi, Ulo, nw);

  dim3 grid(GRU_ROWS / TILE_M, GRU_H / TILE_H);  // (64, 16)
  for (int t = 0; t < GRU_T; ++t) {
    // even t: read out, write ws; odd t: read ws, write out -> t=15 ends in out
    const float* hin = (t & 1) ? hws : out;
    float*       ho  = (t & 1) ? out : hws;
    skipgru_step<<<grid, 256, 0, stream>>>(x, Whi, Wlo, Uhi, Ulo, b, hin, ho, t);
  }
}
